// NT_Xent_7189775253513
// MI455X (gfx1250) — matrix operand order repaired and checked by reading, not compiled, not run on hardware
//
#include <hip/hip_runtime.h>
#include <hip/hip_bf16.h>

// NT-Xent loss, fused WMMA GEMM + streaming log-sum-exp for MI455X (gfx1250).
//
//   B_SZ = 4096, D = 128, N = 8192, TEMP = 0.5  ->  INV_TEMP = 2.0
//   loss = (1/N) * sum_i [ log( sum_{j != i} exp(zn_i.zn_j/T) ) - zn_i.zn_lbl(i)/T ]
//
// Logits are bounded in [-2,2]: no max pass needed, exp cannot overflow or
// hit the subnormal range, so raw v_exp_f32 (no denorm fix-up) is exact here.
// The inner loop is branch-free: it sums exp over ALL columns (including the
// diagonal); the self term exp(2*zn_i.zn_i) and the positive logit are
// computed once per row in the epilogue from the LDS A-tile and folded in.

typedef __attribute__((ext_vector_type(16))) _Float16 v16h;
typedef __attribute__((ext_vector_type(8)))  _Float16 v8h;
typedef __attribute__((ext_vector_type(8)))  float    v8f;

#define NTX_B   4096
#define NTX_D   128
#define NTX_N   8192
#define NTX_INV_TEMP 2.0f
// 2 * log2(e): exp(2a) == exp2(a * NTX_EXP2_SCALE)
#define NTX_EXP2_SCALE 2.885390081777926f

// Fast exp(2a): single v_mul + v_exp_f32 (args bounded, no fix-up needed).
__device__ __forceinline__ float ntx_exp_2x(float a) {
#if __has_builtin(__builtin_amdgcn_exp2f)
    return __builtin_amdgcn_exp2f(a * NTX_EXP2_SCALE);
#else
    return __expf(a * NTX_INV_TEMP);
#endif
}

// ---------------------------------------------------------------------------
// Kernel 1: row-normalize z = concat(z_i, z_j) into f16 workspace.
// One block (128 threads = 4 waves) per row.
// ---------------------------------------------------------------------------
__global__ void ntx_normalize(const float* __restrict__ zi,
                              const float* __restrict__ zj,
                              _Float16* __restrict__ zn) {
    const int row = blockIdx.x;
    const int t   = threadIdx.x;

    float v = (row < NTX_B) ? zi[(size_t)row * NTX_D + t]
                            : zj[(size_t)(row - NTX_B) * NTX_D + t];
    float ss = v * v;
    ss += __shfl_xor(ss, 16, 32);
    ss += __shfl_xor(ss,  8, 32);
    ss += __shfl_xor(ss,  4, 32);
    ss += __shfl_xor(ss,  2, 32);
    ss += __shfl_xor(ss,  1, 32);

    __shared__ float w[4];
    if ((t & 31) == 0) w[t >> 5] = ss;
    __syncthreads();
    const float tot  = w[0] + w[1] + w[2] + w[3];
    const float rinv = 1.0f / fmaxf(sqrtf(tot), 1e-8f);
    zn[(size_t)row * NTX_D + t] = (_Float16)(v * rinv);
}

// ---------------------------------------------------------------------------
// Kernel 2: fused sim-GEMM + exp row-sum. One block (256 threads = 8 waves)
// per 16 output rows; each wave owns a 16-column strip and sweeps all 8192
// columns with v_wmma_f32_16x16x32_f16 (K = 128 in 4 steps).
// ---------------------------------------------------------------------------
__global__ void ntx_simloss(const _Float16* __restrict__ zn,
                            float* __restrict__ partials) {
    const int tid  = threadIdx.x;
    const int lane = tid & 31;
    const int wave = tid >> 5;
    const int half = lane >> 4;   // 0: lanes 0-15, 1: lanes 16-31
    const int l16  = lane & 15;
    const int rowbase = blockIdx.x * 16;

    __shared__ __align__(16) _Float16 lds_a[16 * NTX_D]; // 4 KB A tile
    __shared__ float blocksum[16];
    __shared__ float posv[16];
    __shared__ float selfv[16];
    __shared__ float red[16];

    // Stage the 16x128 f16 A tile (16 B per thread, row-major preserved).
    *(v8h*)&lds_a[tid * 8] =
        *(const v8h*)&zn[(size_t)rowbase * NTX_D + tid * 8];
    if (tid < 16) blocksum[tid] = 0.0f;
    __syncthreads();

    // Build the 4 loop-invariant A fragments (16x32 f16 each).
    // ISA 16-bit A layout: lane (l16) = row M; lane-half selects K 0-7/16-23
    // vs 8-15/24-31 -> two contiguous 16 B chunks per fragment.
    v16h afrag[4];
    {
        const int kb = half * 8;
        #pragma unroll
        for (int ks = 0; ks < 4; ++ks) {
            const int k0 = ks * 32;
            v8h a0 = *(const v8h*)&lds_a[l16 * NTX_D + k0 + kb];
            v8h a1 = *(const v8h*)&lds_a[l16 * NTX_D + k0 + kb + 16];
            afrag[ks] = __builtin_shufflevector(a0, a1,
                0, 1, 2, 3, 4, 5, 6, 7, 8, 9, 10, 11, 12, 13, 14, 15);
        }
    }

    float rs[8] = {0.f, 0.f, 0.f, 0.f, 0.f, 0.f, 0.f, 0.f};
    const int kstart = half * 8;   // B layout: lane-half selects K 0-7,16-23/8-15,24-31

    // Branch-free hot loop: 4 WMMA + 8 (mul, exp2, add) per 16x128 strip.
    // Unroll 2x so the compiler software-pipelines loads of the next strip
    // under the WMMA + exp work of the current one.
    #pragma unroll 2
    for (int ct = 0; ct < NTX_N / 128; ++ct) {
        const int gcol = ct * 128 + wave * 16 + l16;
        const _Float16* bp = zn + (size_t)gcol * NTX_D + kstart;

        // Speculative prefetch of next strip (wraps on last iter; harmless).
        __builtin_prefetch(
            zn + (size_t)((gcol + 128) & (NTX_N - 1)) * NTX_D, 0, 1);

        v8f acc = {};
        #pragma unroll
        for (int ks = 0; ks < 4; ++ks) {
            v8h b0 = *(const v8h*)(bp + ks * 32);
            v8h b1 = *(const v8h*)(bp + ks * 32 + 16);
            v16h bfrag = __builtin_shufflevector(b0, b1, // two 16 B chunks
                0, 1, 2, 3, 4, 5, 6, 7, 8, 9, 10, 11, 12, 13, 14, 15);
            acc = __builtin_amdgcn_wmma_f32_16x16x32_f16(
                false, afrag[ks], false, bfrag, (short)0, acc, false, false);
        }
        #pragma unroll
        for (int e = 0; e < 8; ++e)
            rs[e] += ntx_exp_2x(acc[e]);
    }

    // Reduce row sums across the 16 lanes of each half, then across waves.
    #pragma unroll
    for (int e = 0; e < 8; ++e) {
        float v = rs[e];
        v += __shfl_xor(v, 1, 32);
        v += __shfl_xor(v, 2, 32);
        v += __shfl_xor(v, 4, 32);
        v += __shfl_xor(v, 8, 32);
        if (l16 == 0) atomicAdd(&blocksum[e + half * 8], v); // ds_add_f32
    }

    // Per-row positive logit and self dot (16 rows x 16 threads each, 8
    // elements per thread) from the LDS A-tile + the label row in L2.
    {
        const int r    = tid >> 4;       // row 0..15
        const int c    = tid & 15;       // 16 threads per row
        const int grow = rowbase + r;
        const int lbl  = (grow < NTX_B) ? grow + NTX_B : grow - NTX_B;
        const _Float16* lp = zn + (size_t)lbl * NTX_D + c * 8;
        float p = 0.0f, sd = 0.0f;
        #pragma unroll
        for (int k = 0; k < 8; ++k) {
            const float a = (float)lds_a[r * NTX_D + c * 8 + k];
            p  = fmaf(a, (float)lp[k], p);
            sd = fmaf(a, a, sd);
        }
        p  += __shfl_xor(p,  1, 32); sd += __shfl_xor(sd, 1, 32);
        p  += __shfl_xor(p,  2, 32); sd += __shfl_xor(sd, 2, 32);
        p  += __shfl_xor(p,  4, 32); sd += __shfl_xor(sd, 4, 32);
        p  += __shfl_xor(p,  8, 32); sd += __shfl_xor(sd, 8, 32);
        if (c == 0) {
            posv[r]  = p  * NTX_INV_TEMP;
            selfv[r] = sd * NTX_INV_TEMP;
        }
    }
    __syncthreads();

    // Fold out the diagonal term and form log(S_i) - pos_i.
    if (tid < 16)
        red[tid] = __logf(blocksum[tid] - ntx_exp_2x(selfv[tid] * 0.5f))
                   - posv[tid];
    __syncthreads();
    if (tid == 0) {
        float s = 0.0f;
        #pragma unroll
        for (int i = 0; i < 16; ++i) s += red[i];
        partials[blockIdx.x] = s;
    }
}

// ---------------------------------------------------------------------------
// Kernel 3: reduce 512 block partials -> scalar loss.
// ---------------------------------------------------------------------------
__global__ void ntx_finalize(const float* __restrict__ partials,
                             float* __restrict__ out) {
    const int tid = threadIdx.x; // 256 threads
    float v = partials[tid] + partials[tid + 256];
    v += __shfl_xor(v, 16, 32);
    v += __shfl_xor(v,  8, 32);
    v += __shfl_xor(v,  4, 32);
    v += __shfl_xor(v,  2, 32);
    v += __shfl_xor(v,  1, 32);

    __shared__ float w[8];
    if ((tid & 31) == 0) w[tid >> 5] = v;
    __syncthreads();
    if (tid == 0) {
        float s = 0.0f;
        #pragma unroll
        for (int i = 0; i < 8; ++i) s += w[i];
        out[0] = s * (1.0f / (float)NTX_N);
    }
}

// ---------------------------------------------------------------------------
extern "C" void kernel_launch(void* const* d_in, const int* in_sizes, int n_in,
                              void* d_out, int out_size, void* d_ws, size_t ws_size,
                              hipStream_t stream) {
    const float* zi = (const float*)d_in[0];
    const float* zj = (const float*)d_in[1];

    // Workspace: [0, 2 MB) f16 normalized embeddings, then 512 float partials.
    _Float16* zn      = (_Float16*)d_ws;
    float*    partial = (float*)((char*)d_ws + (size_t)NTX_N * NTX_D * sizeof(_Float16));
    float*    out     = (float*)d_out;

    ntx_normalize<<<NTX_N, NTX_D, 0, stream>>>(zi, zj, zn);
    ntx_simloss<<<NTX_N / 16, 256, 0, stream>>>(zn, partial);
    ntx_finalize<<<1, 256, 0, stream>>>(partial, out);
}
